// VectorQuantizerEMA_50276887167269
// MI455X (gfx1250) — compile-verified, matrix operand order checked by reading, not run on hardware
//
#include <hip/hip_runtime.h>
#include <hip/hip_bf16.h>

// ---------------------------------------------------------------------------
// VQ-VAE EMA vector quantizer for MI455X (gfx1250, wave32, WMMA)
//   z: [131072, 256] f32, codebook: [1024, 256] f32
//   dist argmin via V_WMMA_F32_16X16X32_BF16 (||z||^2 dropped: row-constant)
// ---------------------------------------------------------------------------

typedef __bf16        v16bf __attribute__((ext_vector_type(16)));
typedef float         v8f   __attribute__((ext_vector_type(8)));
typedef unsigned int  v4u   __attribute__((ext_vector_type(4)));

union Frag16 { v16bf bf; v4u u[2]; };   // 32 bytes: one 16-bit WMMA operand/lane

#define KCB   1024
#define DDIM  256
#define NROWS (64 * 2048)
#define MTILE 128          // rows per block (8 waves x 16 rows)

// ---------------------------------------------------------------------------
// Kernel 1: codebook -> bf16, ||c||^2, zero accumulators
// grid = 1024 blocks (one codebook row), 256 threads
// ---------------------------------------------------------------------------
__global__ __launch_bounds__(256) void vq_prep(
    const float* __restrict__ cb, __bf16* __restrict__ cb16,
    float* __restrict__ cnorm, float* __restrict__ cnt,
    float* __restrict__ esum, float* __restrict__ loss,
    float* __restrict__ nsum)
{
    __shared__ float red[8];
    const int k = blockIdx.x;
    const int t = threadIdx.x;
    const int i = k * DDIM + t;

    float v = cb[i];
    cb16[i] = (__bf16)v;
    esum[i] = 0.0f;

    float s = v * v;
    #pragma unroll
    for (int off = 16; off > 0; off >>= 1) s += __shfl_xor(s, off, 32);
    if ((t & 31) == 0) red[t >> 5] = s;
    __syncthreads();
    if (t == 0) {
        float tot = 0.0f;
        #pragma unroll
        for (int j = 0; j < 8; ++j) tot += red[j];
        cnorm[k] = tot;
        cnt[k]   = 0.0f;
        if (k == 0) { *loss = 0.0f; *nsum = 0.0f; }
    }
}

// ---------------------------------------------------------------------------
// Kernel 2: WMMA distance GEMM + argmin + gather + segment sums + loss
// grid = NROWS/MTILE = 1024 blocks, 256 threads (8 waves, 16 rows/wave)
// ---------------------------------------------------------------------------
__global__ __launch_bounds__(256) void vq_main(
    const float* __restrict__ z, const float* __restrict__ cb,
    const __bf16* __restrict__ cb16, const float* __restrict__ cnorm,
    float* __restrict__ out_zq, float* __restrict__ out_tok,
    float* __restrict__ cnt, float* __restrict__ esum,
    float* __restrict__ loss)
{
    __shared__ __align__(16) __bf16 lds_z[MTILE * DDIM];   // 64 KB

    const int  tid     = threadIdx.x;
    const long rowBase = (long)blockIdx.x * MTILE;

    // ---- stage z tile as bf16 in LDS (coalesced) ----
    {
        const float* zsrc = z + rowBase * DDIM;
        for (int i = tid; i < MTILE * DDIM; i += 256)
            lds_z[i] = (__bf16)zsrc[i];
    }
    __syncthreads();

    const int wave  = tid >> 5;
    const int lane  = tid & 31;
    const int lhalf = lane >> 4;    // 0: lanes 0-15, 1: lanes 16-31
    const int l16   = lane & 15;
    const int mbase = wave * 16;    // this wave's 16 rows within the tile

    // ---- preload A fragments (full K = 256 -> 8 chunks of 32) ----
    // 16-bit A layout (ISA 7.12.2): lane holds row m = l16; halves at k and k+16,
    // high half-wave shifted by +8 in K.
    Frag16 a[8];
    {
        const __bf16* zr = lds_z + (mbase + l16) * DDIM;
        #pragma unroll
        for (int c = 0; c < 8; ++c) {
            const int kb = c * 32 + lhalf * 8;
            a[c].u[0] = *(const v4u*)(zr + kb);
            a[c].u[1] = *(const v4u*)(zr + kb + 16);
        }
    }

    float best[8];
    int   bidx[8];
    #pragma unroll
    for (int r = 0; r < 8; ++r) { best[r] = 3.4e38f; bidx[r] = 0; }

    // ---- sweep codebook in 16-column tiles: dist = ||c||^2 - 2 z.c ----
    for (int nt = 0; nt < KCB / 16; ++nt) {
        const __bf16* brow = cb16 + (nt * 16 + l16) * DDIM;
        if (nt + 1 < KCB / 16)
            __builtin_prefetch(cb16 + ((nt + 1) * 16 + l16) * DDIM, 0, 1);

        v8f acc = {};
        #pragma unroll
        for (int c = 0; c < 8; ++c) {
            Frag16 b;
            const int kb = c * 32 + lhalf * 8;
            b.u[0] = *(const v4u*)(brow + kb);
            b.u[1] = *(const v4u*)(brow + kb + 16);
            acc = __builtin_amdgcn_wmma_f32_16x16x32_bf16(
                      false, a[c].bf, false, b.bf, (short)0, acc, false, false);
        }

        const int   nidx = nt * 16 + l16;
        const float cn   = cnorm[nidx];
        #pragma unroll
        for (int r = 0; r < 8; ++r) {
            const float d = cn - 2.0f * acc[r];
            if (d < best[r]) { best[r] = d; bidx[r] = nidx; }
        }
    }

    // ---- argmin across the 16 lanes of each half-wave (butterfly) ----
    #pragma unroll
    for (int r = 0; r < 8; ++r) {
        #pragma unroll
        for (int off = 1; off < 16; off <<= 1) {
            const float ob = __shfl_xor(best[r], off, 32);
            const int   oi = __shfl_xor(bidx[r], off, 32);
            if (ob < best[r] || (ob == best[r] && oi < bidx[r])) {
                best[r] = ob; bidx[r] = oi;
            }
        }
    }

    // ---- tokens (lane 0 -> rows 0..7, lane 16 -> rows 8..15) ----
    if (l16 == 0) {
        #pragma unroll
        for (int r = 0; r < 8; ++r) {
            const long row = rowBase + mbase + r + lhalf * 8;
            out_tok[row] = (float)bidx[r];
        }
    }

    // ---- gather z_q, commit-loss partial, EMA segment sums ----
    float lacc = 0.0f;
    #pragma unroll
    for (int m = 0; m < 16; ++m) {
        const int  idx  = __shfl(bidx[m & 7], (m >> 3) * 16, 32);
        const long row  = rowBase + mbase + m;
        const float* crow = cb     + (long)idx * DDIM;
        const float* zrow = z      + row * DDIM;
        float*       orow = out_zq + row * DDIM;
        float*       erow = esum   + (long)idx * DDIM;
        for (int i = lane; i < DDIM; i += 32) {
            const float cq = crow[i];
            const float zv = zrow[i];
            orow[i] = cq;                        // z_q_st == z_q numerically
            const float df = cq - zv;
            lacc += df * df;
            atomicAdd(&erow[i], zv);             // embed_sum segment add
        }
        if (lane == 0) atomicAdd(&cnt[idx], 1.0f);
    }

    #pragma unroll
    for (int off = 16; off > 0; off >>= 1) lacc += __shfl_xor(lacc, off, 32);
    if (lane == 0) atomicAdd(loss, lacc);
}

// ---------------------------------------------------------------------------
// Kernel 3: EMA cluster sizes, n = sum, commit loss scalar (1 block x 1024)
// ---------------------------------------------------------------------------
__global__ __launch_bounds__(1024) void vq_ema_size(
    const float* __restrict__ ecs, const float* __restrict__ cnt,
    const float* __restrict__ loss, float* __restrict__ out_necs,
    float* __restrict__ out_loss, float* __restrict__ nsum)
{
    __shared__ float red[32];
    const int k = threadIdx.x;
    const float necs = 0.95f * ecs[k] + 0.05f * cnt[k];
    out_necs[k] = necs;

    float s = necs;
    #pragma unroll
    for (int off = 16; off > 0; off >>= 1) s += __shfl_xor(s, off, 32);
    if ((k & 31) == 0) red[k >> 5] = s;
    __syncthreads();
    if (k < 32) {
        float v = red[k];
        #pragma unroll
        for (int off = 16; off > 0; off >>= 1) v += __shfl_xor(v, off, 32);
        if (k == 0) {
            *nsum = v;
            *out_loss = 0.5f * (*loss) / (float)((long)NROWS * DDIM);
        }
    }
}

// ---------------------------------------------------------------------------
// Kernel 4: EMA codebook + smoothed normalization (1024 blocks x 256)
// ---------------------------------------------------------------------------
__global__ __launch_bounds__(256) void vq_ema_cb(
    const float* __restrict__ ecb, const float* __restrict__ esum,
    const float* __restrict__ necs, const float* __restrict__ nsum,
    float* __restrict__ out_ncb, float* __restrict__ out_necb)
{
    const int k = blockIdx.x;
    const int t = threadIdx.x;
    const long i = (long)k * DDIM + t;
    const float necb = 0.95f * ecb[i] + 0.05f * esum[i];
    out_necb[i] = necb;
    const float n  = *nsum;
    const float sm = (necs[k] + 1e-5f) / (n + (float)KCB * 1e-5f) * n;
    out_ncb[i] = necb / sm;
}

// ---------------------------------------------------------------------------
extern "C" void kernel_launch(void* const* d_in, const int* in_sizes, int n_in,
                              void* d_out, int out_size, void* d_ws, size_t ws_size,
                              hipStream_t stream)
{
    const float* z   = (const float*)d_in[0];   // 64*2048*256
    const float* cb  = (const float*)d_in[1];   // 1024*256
    const float* ecs = (const float*)d_in[2];   // 1024
    const float* ecb = (const float*)d_in[3];   // 1024*256

    // workspace carve-up
    char* w = (char*)d_ws;
    __bf16* cb16  = (__bf16*)w;                         //  524288 B
    float*  cnorm = (float*)(w + 524288);               //    4096 B
    float*  cnt   = cnorm + KCB;                        //    4096 B
    float*  loss  = cnt + KCB;                          //       4 B
    float*  nsum  = loss + 1;                           //       4 B
    float*  esum  = (float*)(w + 532512);               // 1048576 B (aligned)

    // output carve-up (tuple concatenated flat, f32)
    float* o = (float*)d_out;
    float* out_zq   = o;                                 // 33554432
    float* out_tok  = out_zq  + (long)NROWS * DDIM;      //   131072
    float* out_loss = out_tok + NROWS;                   //        1
    float* out_ncb  = out_loss + 1;                      //   262144
    float* out_necs = out_ncb + (long)KCB * DDIM;        //     1024
    float* out_necb = out_necs + KCB;                    //   262144

    vq_prep<<<KCB, 256, 0, stream>>>(cb, cb16, cnorm, cnt, esum, loss, nsum);
    vq_main<<<NROWS / MTILE, 256, 0, stream>>>(z, cb, cb16, cnorm,
                                               out_zq, out_tok, cnt, esum, loss);
    vq_ema_size<<<1, 1024, 0, stream>>>(ecs, cnt, loss, out_necs, out_loss, nsum);
    vq_ema_cb<<<KCB, 256, 0, stream>>>(ecb, esum, out_necs, nsum, out_ncb, out_necb);
}
